// FeatureFusion_29497835389413
// MI455X (gfx1250) — matrix operand order repaired and checked by reading, not compiled, not run on hardware
//
#include <hip/hip_runtime.h>

#define OSZ   14
#define NODES 196          // 14*14
#define KROI  256
#define NB    4
#define CCH   256
#define HH    112
#define WW    112
#define PPIX  256          // 16x16 zero-bordered pixel tile
#define NPAD  224          // nodes padded to multiple of 32 (K dim of GCN GEMMs)
#define KT_CONV 72         // 2304 / 32
#define KT_GCN  7          // 224 / 32

typedef __bf16 bf16_t;
typedef __attribute__((ext_vector_type(16))) __bf16 v16bf;
typedef __attribute__((ext_vector_type(8)))  float  v8f;

static __device__ __forceinline__ v8f wmma_bf16(v16bf a, v16bf b, v8f c) {
  return __builtin_amdgcn_wmma_f32_16x16x32_bf16(
      false, a, false, b, (short)0, c, false, false);
}

// Fragment-major packed A: [mtile][ktile][lane][16] -> one aligned 32B load/lane.
static __device__ __forceinline__ v16bf load_a_pk(const bf16_t* __restrict__ Apk,
                                                  int ktiles, int mt, int kt) {
  const int lane = threadIdx.x & 31;
  return *(const v16bf*)(Apk + (size_t)(((mt * ktiles) + kt) * 32 + lane) * 16);
}

// K index inside a 16x32 bf16 A fragment (contiguous order k = 16h + e): element e of lane half.
static __device__ __forceinline__ int a_frag_k(int e, int khalf) {
  return khalf + e;
}

// =====================================================================
// Pack / elementwise kernels
// =====================================================================

// conv weights (64,256,3,3) fp32 -> fragment-major bf16 with tap-major K
// (kk = tap*256 + c), layout [4 mtiles][72 ktiles][32 lanes][16]
__global__ void pack_wconv_kernel(const float* __restrict__ w,
                                  bf16_t* __restrict__ wpk) {
  int i = blockIdx.x * blockDim.x + threadIdx.x;
  const int total = 4 * KT_CONV * 512;
  if (i >= total) return;
  const int e = i & 15, lane = (i >> 4) & 31;
  const int kt = (i >> 9) % KT_CONV, mt = (i >> 9) / KT_CONV;
  const int m = mt * 16 + (lane & 15);
  const int kk = kt * 32 + a_frag_k(e, (lane & 16) ? 16 : 0);
  const int t = kk >> 8, c = kk & 255;
  wpk[i] = (bf16_t)w[(m * CCH + c) * 9 + t];
}

// aff weights (196,64) fp32 -> [13 mtiles][2 ktiles][32][16], zero-fill m>=196
__global__ void pack_waff_kernel(const float* __restrict__ w,
                                 bf16_t* __restrict__ wpk) {
  int i = blockIdx.x * blockDim.x + threadIdx.x;
  const int total = 13 * 2 * 512;
  if (i >= total) return;
  const int e = i & 15, lane = (i >> 4) & 31;
  const int kt = (i >> 9) & 1, mt = (i >> 9) >> 1;
  const int m = mt * 16 + (lane & 15);
  const int kk = kt * 32 + a_frag_k(e, (lane & 16) ? 16 : 0);   // < 64
  wpk[i] = (bf16_t)((m < NODES) ? w[m * 64 + kk] : 0.0f);
}

// Normalized AsT[j][i] = dis[i]*Aff[i][j]*dis[j], packed fragment-major per ROI:
// [roi][13 mtiles][7 ktiles][32][16], zero-fill beyond 196.
__global__ void pack_ast_kernel(const float* __restrict__ Aff,
                                const float* __restrict__ dis,
                                bf16_t* __restrict__ pk) {
  int i = blockIdx.x * blockDim.x + threadIdx.x;
  const int per_roi = 13 * KT_GCN * 512;
  const int total = KROI * per_roi;
  if (i >= total) return;
  const int roi = i / per_roi;
  const int rem = i - roi * per_roi;
  const int e = rem & 15, lane = (rem >> 4) & 31;
  const int kt = (rem >> 9) % KT_GCN, mt = (rem >> 9) / KT_GCN;
  const int j  = mt * 16 + (lane & 15);
  const int ii = kt * 32 + a_frag_k(e, (lane & 16) ? 16 : 0);
  float val = 0.0f;
  if (j < NODES && ii < NODES)
    val = dis[roi * NODES + ii] * Aff[(size_t)roi * (NODES * NODES) + ii * NODES + j] *
          dis[roi * NODES + j];
  pk[i] = (bf16_t)val;
}

// BN+ReLU + transpose to pixel-major with 1-pixel zero border:
// in  [roi][C][196] fp32 ;  out [roi][256 ppix (16x16)][256 ch] bf16
__global__ void bnrelu_tpad_kernel(const float* __restrict__ x,
                                   const float* __restrict__ g,
                                   const float* __restrict__ b,
                                   const float* __restrict__ m,
                                   const float* __restrict__ v,
                                   bf16_t* __restrict__ outT) {
  int i = blockIdx.x * blockDim.x + threadIdx.x;
  const int total = KROI * PPIX * CCH;
  if (i >= total) return;
  const int c = i & 255, ppix = (i >> 8) & 255, roi = i >> 16;
  const int yy = ppix >> 4, xx = ppix & 15;
  float val = 0.0f;
  if (yy >= 1 && yy <= OSZ && xx >= 1 && xx <= OSZ) {
    const int p = (yy - 1) * OSZ + (xx - 1);
    const float s = rsqrtf(v[c] + 1e-5f);
    val = fmaxf((x[(size_t)roi * (CCH * NODES) + c * NODES + p] - m[c]) * s * g[c] + b[c],
                0.0f);
  }
  outT[i] = (bf16_t)val;
}

// BN+ReLU of 'a' [roi][64][196] -> transposed, padded: [roi][208][64] bf16
__global__ void bnrelu_aT_kernel(const float* __restrict__ a,
                                 const float* __restrict__ g,
                                 const float* __restrict__ b,
                                 const float* __restrict__ m,
                                 const float* __restrict__ v,
                                 bf16_t* __restrict__ outT) {
  int i = blockIdx.x * blockDim.x + threadIdx.x;
  const int total = KROI * 208 * 64;
  if (i >= total) return;
  const int c = i & 63, n = (i >> 6) % 208, roi = i / (208 * 64);
  float val = 0.0f;
  if (n < NODES) {
    const float s = rsqrtf(v[c] + 1e-5f);
    val = fmaxf((a[(size_t)roi * (64 * NODES) + c * NODES + n] - m[c]) * s * g[c] + b[c],
                0.0f);
  }
  outT[i] = (bf16_t)val;
}

__global__ void roi_align_kernel(const float* __restrict__ sem,
                                 const float* __restrict__ boxes,
                                 float* __restrict__ out) {
  int i = blockIdx.x * blockDim.x + threadIdx.x;
  const int total = KROI * CCH * NODES;
  if (i >= total) return;
  const int k   = i / (CCH * NODES);
  const int rem = i - k * (CCH * NODES);
  const int c   = rem / NODES;
  const int p   = rem - c * NODES;
  const int py  = p / OSZ, px = p % OSZ;

  int bidx = (int)boxes[k * 5 + 0];
  bidx = bidx < 0 ? 0 : (bidx > NB - 1 ? NB - 1 : bidx);
  const float x1b = boxes[k * 5 + 1], y1b = boxes[k * 5 + 2];
  const float x2b = boxes[k * 5 + 3], y2b = boxes[k * 5 + 4];
  const float bw = fmaxf(x2b - x1b, 1.0f) * (1.0f / OSZ);
  const float bh = fmaxf(y2b - y1b, 1.0f) * (1.0f / OSZ);
  const float* base = sem + (size_t)(bidx * CCH + c) * (HH * WW);

  float acc = 0.0f;
#pragma unroll
  for (int gy = 0; gy < 2; ++gy) {
#pragma unroll
    for (int gx = 0; gx < 2; ++gx) {
      const float yc = y1b + bh * (((float)(py * 2 + gy) + 0.5f) * 0.5f);
      const float xc = x1b + bw * (((float)(px * 2 + gx) + 0.5f) * 0.5f);
      const bool my = (yc >= -1.0f) && (yc <= (float)HH);
      const bool mx = (xc >= -1.0f) && (xc <= (float)WW);
      const float ycc = fminf(fmaxf(yc, 0.0f), (float)(HH - 1));
      const float xcc = fminf(fmaxf(xc, 0.0f), (float)(WW - 1));
      int yl = (int)floorf(ycc); int yh = yl + 1; if (yh > HH - 1) yh = HH - 1;
      int xl = (int)floorf(xcc); int xh = xl + 1; if (xh > WW - 1) xh = WW - 1;
      const float ly = ycc - (float)yl, lx = xcc - (float)xl;
      const float v00 = base[yl * WW + xl], v01 = base[yl * WW + xh];
      const float v10 = base[yh * WW + xl], v11 = base[yh * WW + xh];
      float val = v00 * (1 - ly) * (1 - lx) + v01 * (1 - ly) * lx +
                  v10 * ly * (1 - lx)       + v11 * ly * lx;
      if (my && mx) acc += val;
    }
  }
  out[i] = acc * 0.25f;
}

__global__ void degdis_kernel(const float* __restrict__ Aff,
                              float* __restrict__ dis) {
  int i = blockIdx.x * blockDim.x + threadIdx.x;
  if (i >= KROI * NODES) return;
  const int k = i / NODES, j = i - k * NODES;
  const float* A = Aff + (size_t)k * (NODES * NODES);
  float s = 0.0f;
  for (int r = 0; r < NODES; ++r) s += A[r * NODES + j];
  dis[i] = rsqrtf(s);
}

// XwT[roi][f(16)][node 224 pad] = sum_c roi_feature[roi][c][node] * gcn1_w[c][f]
__global__ void xwT_kernel(const float* __restrict__ rf,
                           const float* __restrict__ w1,
                           bf16_t* __restrict__ XwT) {
  int i = blockIdx.x * blockDim.x + threadIdx.x;
  const int total = KROI * 16 * NPAD;
  if (i >= total) return;
  const int node = i % NPAD;
  const int f    = (i / NPAD) & 15;
  const int roi  = i / (NPAD * 16);
  float s = 0.0f;
  if (node < NODES) {
    const float* X = rf + (size_t)roi * (CCH * NODES);
    for (int c = 0; c < CCH; ++c) s += X[c * NODES + node] * w1[c * 16 + f];
  }
  XwT[i] = (bf16_t)s;
}

// YT[roi][c(256)][node 224 pad] = sum_f x1[roi][node][f] * gcn2_w[f][c]
__global__ void yT_kernel(const bf16_t* __restrict__ x1,
                          const float* __restrict__ w2,
                          bf16_t* __restrict__ YT) {
  int i = blockIdx.x * blockDim.x + threadIdx.x;
  const int total = KROI * CCH * NPAD;
  if (i >= total) return;
  const int node = i % NPAD;
  const int c    = (i / NPAD) & 255;
  const int roi  = i / (NPAD * CCH);
  float s = 0.0f;
  if (node < NODES) {
    const bf16_t* xr = x1 + (size_t)roi * (NODES * 16) + node * 16;
#pragma unroll
    for (int f = 0; f < 16; ++f) s += (float)xr[f] * w2[f * CCH + c];
  }
  YT[i] = (bf16_t)s;
}

// =====================================================================
// WMMA GEMM kernels — branch-free vector fragment loads
// =====================================================================

// a[roi][64][196] = conv3x3(sem) + conv3x3(fpn) + biases.
// One wave per (roi, ntile); B fragment reused across all 4 M-tiles.
// Tap-outer loop: im2col pixel offset computed once per tap (9x), not per K-step.
__global__ __launch_bounds__(128) void conv3x3_dual_wmma(
    const bf16_t* __restrict__ actT_sem, const bf16_t* __restrict__ actT_fpn,
    const bf16_t* __restrict__ wpk_sem,  const bf16_t* __restrict__ wpk_fpn,
    const float*  __restrict__ b_sem,    const float*  __restrict__ b_fpn,
    float* __restrict__ a_out) {
  const int wave = blockIdx.x * (blockDim.x >> 5) + (threadIdx.x >> 5);
  const int roi  = wave / 13;
  const int nt   = wave - roi * 13;
  if (roi >= KROI) return;
  const int lane = threadIdx.x & 31;
  int pix = nt * 16 + (lane & 15);
  if (pix > NODES - 1) pix = NODES - 1;            // pad columns: masked at store
  const int py = pix / OSZ, px = pix % OSZ;
  const int chalf = (lane & 16) ? 16 : 0;
  const bf16_t* aS = actT_sem + (size_t)roi * (PPIX * CCH) + chalf;
  const bf16_t* aF = actT_fpn + (size_t)roi * (PPIX * CCH) + chalf;

  v8f acc[4] = {};
  for (int t = 0; t < 9; ++t) {                    // tap (K is tap-major)
    const int ty = t / 3, tx = t - 3 * ty;
    const size_t boff = (size_t)((py + ty) * 16 + (px + tx)) * CCH;  // zero border
    const bf16_t* pS = aS + boff;
    const bf16_t* pF = aF + boff;
    __builtin_prefetch(pS + CCH * 16, 0, 1);       // next tap's row
    __builtin_prefetch(pF + CCH * 16, 0, 1);
    for (int kc = 0; kc < 8; ++kc) {               // 8 channel slabs of 32
      const int kt = t * 8 + kc;
      const v16bf bs = *(const v16bf*)(pS + (kc << 5));
#pragma unroll
      for (int mt = 0; mt < 4; ++mt)
        acc[mt] = wmma_bf16(load_a_pk(wpk_sem, KT_CONV, mt, kt), bs, acc[mt]);
      const v16bf bf = *(const v16bf*)(pF + (kc << 5));
#pragma unroll
      for (int mt = 0; mt < 4; ++mt)
        acc[mt] = wmma_bf16(load_a_pk(wpk_fpn, KT_CONV, mt, kt), bf, acc[mt]);
    }
  }
  const int n  = nt * 16 + (lane & 15);
  const int mh = (lane & 16) ? 8 : 0;
  if (n < NODES) {
#pragma unroll
    for (int mt = 0; mt < 4; ++mt)
#pragma unroll
      for (int r = 0; r < 8; ++r) {
        const int m = mt * 16 + mh + r;            // < 64
        a_out[(size_t)roi * (64 * NODES) + m * NODES + n] =
            acc[mt][r] + b_sem[m] + b_fpn[m];
      }
  }
}

// Aff[roi][196][196] = sigmoid( w_aff @ a_act + bias ); one wave per 16x16 tile
__global__ __launch_bounds__(128) void aff_wmma(
    const bf16_t* __restrict__ wpk_aff, const bf16_t* __restrict__ aT,
    const float* __restrict__ aff_b,    float* __restrict__ Aff) {
  const int wave  = blockIdx.x * (blockDim.x >> 5) + (threadIdx.x >> 5);
  const int tiles = 13 * 13;
  const int roi   = wave / tiles;
  const int t     = wave - roi * tiles;
  if (roi >= KROI) return;
  const int mt = t / 13, nt = t % 13;
  const int lane = threadIdx.x & 31;
  const int n  = nt * 16 + (lane & 15);            // < 208 (aT padded+zeroed)
  const int kb = (lane & 16) ? 16 : 0;
  const bf16_t* Bm = aT + (size_t)roi * (208 * 64);
  v8f acc = {};
#pragma unroll
  for (int kt = 0; kt < 2; ++kt) {
    const v16bf a = load_a_pk(wpk_aff, 2, mt, kt);
    const v16bf b = *(const v16bf*)(Bm + (size_t)n * 64 + (kt * 32 + kb));
    acc = wmma_bf16(a, b, acc);
  }
  const int mh = (lane & 16) ? 8 : 0;
#pragma unroll
  for (int r = 0; r < 8; ++r) {
    const int m = mt * 16 + mh + r;
    if (m < NODES && n < NODES) {
      const float v = acc[r] + aff_b[m];
      Aff[(size_t)roi * (NODES * NODES) + m * NODES + n] = 1.0f / (1.0f + __expf(-v));
    }
  }
}

// x1[roi][196][16] = relu( AsT @ Xw + b1 ); one wave per (roi, mtile)
__global__ __launch_bounds__(128) void gcn1_wmma(
    const bf16_t* __restrict__ AsTpk, const bf16_t* __restrict__ XwT,
    const float* __restrict__ b1,     bf16_t* __restrict__ x1) {
  const int wave = blockIdx.x * (blockDim.x >> 5) + (threadIdx.x >> 5);
  const int roi  = wave / 13;
  const int mt   = wave - roi * 13;
  if (roi >= KROI) return;
  const int lane = threadIdx.x & 31;
  const int nf   = lane & 15;                      // f index
  const int kb   = (lane & 16) ? 16 : 0;
  const bf16_t* Apk = AsTpk + (size_t)roi * (13 * KT_GCN * 512);
  const bf16_t* Xr  = XwT   + (size_t)roi * (16 * NPAD);
  v8f acc = {};
  for (int kt = 0; kt < KT_GCN; ++kt) {
    const v16bf a = load_a_pk(Apk, KT_GCN, mt, kt);
    const v16bf b = *(const v16bf*)(Xr + (size_t)nf * NPAD + (kt * 32 + kb));
    acc = wmma_bf16(a, b, acc);
  }
  const int mh = (lane & 16) ? 8 : 0;
#pragma unroll
  for (int r = 0; r < 8; ++r) {
    const int m = mt * 16 + mh + r;
    if (m < NODES) {
      const float v = fmaxf(acc[r] + b1[nf], 0.0f);
      x1[(size_t)roi * (NODES * 16) + m * 16 + nf] = (bf16_t)v;
    }
  }
}

// out[roi][c][node] += (AsT @ Y)[node][c] + b2[c]
// One block (4 waves) per (roi, mtile); the shared 7KB AsT panel is staged
// once in LDS and fragment-read via ds_load_b128; each wave owns 64 channels.
__global__ __launch_bounds__(128) void gcn2_wmma(
    const bf16_t* __restrict__ AsTpk, const bf16_t* __restrict__ YT,
    const float* __restrict__ b2,     float* __restrict__ out) {
  const int roi = blockIdx.x / 13;
  const int mt  = blockIdx.x - roi * 13;
  const int ng  = threadIdx.x >> 5;                // wave id = channel group
  const int lane = threadIdx.x & 31;

  __shared__ __align__(32) bf16_t smA[KT_GCN * 512];   // 7 ktiles * 32 lanes * 16

  // cooperative stage of this mtile's packed A panel (7168 B, b64 chunks)
  {
    const uint64_t* src =
        (const uint64_t*)(AsTpk + (size_t)(roi * 13 + mt) * (KT_GCN * 512));
    uint64_t* dst = (uint64_t*)smA;
#pragma unroll
    for (int q = 0; q < 7; ++q)                    // 896 qwords / 128 threads
      dst[threadIdx.x + q * 128] = src[threadIdx.x + q * 128];
  }
  __syncthreads();

  const int kb = (lane & 16) ? 16 : 0;
  const bf16_t* Yr = YT + (size_t)roi * (CCH * NPAD);
  v8f acc[4] = {};
  for (int kt = 0; kt < KT_GCN; ++kt) {
    const v16bf a = *(const v16bf*)(smA + (kt * 32 + lane) * 16);   // ds_load
    const int kbase = kt * 32 + kb;
#pragma unroll
    for (int q = 0; q < 4; ++q) {
      const int n = ng * 64 + q * 16 + (lane & 15);
      const v16bf b = *(const v16bf*)(Yr + (size_t)n * NPAD + kbase);
      acc[q] = wmma_bf16(a, b, acc[q]);
    }
  }
  const int mh = (lane & 16) ? 8 : 0;
#pragma unroll
  for (int q = 0; q < 4; ++q) {
    const int n = ng * 64 + q * 16 + (lane & 15);  // channel, < 256
    const float bias = b2[n];
#pragma unroll
    for (int r = 0; r < 8; ++r) {
      const int m = mt * 16 + mh + r;              // node
      if (m < NODES) {
        const size_t idx = (size_t)roi * (CCH * NODES) + (size_t)n * NODES + m;
        out[idx] = out[idx] + acc[q][r] + bias;    // residual add onto roi_sem
      }
    }
  }
}

// =====================================================================
extern "C" void kernel_launch(void* const* d_in, const int* in_sizes, int n_in,
                              void* d_out, int out_size, void* d_ws, size_t ws_size,
                              hipStream_t stream) {
  (void)in_sizes; (void)n_in; (void)out_size; (void)ws_size;
  const float* roi_feature = (const float*)d_in[0];
  const float* semantic    = (const float*)d_in[1];
  const float* boxes       = (const float*)d_in[2];
  const float* bn_sem_g = (const float*)d_in[3];
  const float* bn_sem_b = (const float*)d_in[4];
  const float* bn_sem_m = (const float*)d_in[5];
  const float* bn_sem_v = (const float*)d_in[6];
  const float* conv_sem_w = (const float*)d_in[7];
  const float* conv_sem_b = (const float*)d_in[8];
  const float* bn_fpn_g = (const float*)d_in[9];
  const float* bn_fpn_b = (const float*)d_in[10];
  const float* bn_fpn_m = (const float*)d_in[11];
  const float* bn_fpn_v = (const float*)d_in[12];
  const float* conv_fpn_w = (const float*)d_in[13];
  const float* conv_fpn_b = (const float*)d_in[14];
  const float* bn_aff_g = (const float*)d_in[15];
  const float* bn_aff_b = (const float*)d_in[16];
  const float* bn_aff_m = (const float*)d_in[17];
  const float* bn_aff_v = (const float*)d_in[18];
  const float* conv_aff_w = (const float*)d_in[19];
  const float* conv_aff_b = (const float*)d_in[20];
  const float* gcn1_w = (const float*)d_in[21];
  const float* gcn1_b = (const float*)d_in[22];
  const float* gcn2_w = (const float*)d_in[23];
  const float* gcn2_b = (const float*)d_in[24];
  float* out = (float*)d_out;

  // ---- workspace carve-out (~155 MB)
  char* wsb = (char*)d_ws;
  size_t off = 0;
  auto walloc = [&](size_t bytes) -> void* {
    void* r = wsb + off;
    off += (bytes + 255) & ~(size_t)255;
    return r;
  };
  bf16_t* actT_sem = (bf16_t*)walloc((size_t)KROI * PPIX * CCH * 2);
  bf16_t* actT_fpn = (bf16_t*)walloc((size_t)KROI * PPIX * CCH * 2);
  bf16_t* wpk_sem  = (bf16_t*)walloc((size_t)4 * KT_CONV * 512 * 2);
  bf16_t* wpk_fpn  = (bf16_t*)walloc((size_t)4 * KT_CONV * 512 * 2);
  bf16_t* wpk_aff  = (bf16_t*)walloc((size_t)13 * 2 * 512 * 2);
  float*  a_f32    = (float*) walloc((size_t)KROI * 64 * NODES * 4);
  bf16_t* a_actT   = (bf16_t*)walloc((size_t)KROI * 208 * 64 * 2);
  float*  Aff      = (float*) walloc((size_t)KROI * NODES * NODES * 4);
  float*  dis      = (float*) walloc((size_t)KROI * NODES * 4);
  bf16_t* AsTpk    = (bf16_t*)walloc((size_t)KROI * 13 * KT_GCN * 512 * 2);
  bf16_t* XwT      = (bf16_t*)walloc((size_t)KROI * 16 * NPAD * 2);
  bf16_t* x1b      = (bf16_t*)walloc((size_t)KROI * NODES * 16 * 2);
  bf16_t* YT       = (bf16_t*)Aff;   // Aff (39.3MB) dead after AsTpk built; YT needs 29.4MB

  const int TPB = 256;
  const int tot_out = KROI * CCH * NODES;

  // 1) weight packs (fp32 -> fragment-major bf16)
  pack_wconv_kernel<<<(4 * KT_CONV * 512 + TPB - 1) / TPB, TPB, 0, stream>>>(conv_sem_w, wpk_sem);
  pack_wconv_kernel<<<(4 * KT_CONV * 512 + TPB - 1) / TPB, TPB, 0, stream>>>(conv_fpn_w, wpk_fpn);
  pack_waff_kernel<<<(13 * 2 * 512 + TPB - 1) / TPB, TPB, 0, stream>>>(conv_aff_w, wpk_aff);

  // 2) ROI align -> d_out (residual base)
  roi_align_kernel<<<(tot_out + TPB - 1) / TPB, TPB, 0, stream>>>(semantic, boxes, out);

  // 3) BN+ReLU + transpose to zero-bordered pixel-major bf16
  const int tot_tp = KROI * PPIX * CCH;
  bnrelu_tpad_kernel<<<(tot_tp + TPB - 1) / TPB, TPB, 0, stream>>>(
      out, bn_sem_g, bn_sem_b, bn_sem_m, bn_sem_v, actT_sem);
  bnrelu_tpad_kernel<<<(tot_tp + TPB - 1) / TPB, TPB, 0, stream>>>(
      roi_feature, bn_fpn_g, bn_fpn_b, bn_fpn_m, bn_fpn_v, actT_fpn);

  // 4) dual 3x3 conv: 256 rois * 13 ntiles waves, 4 waves/block
  conv3x3_dual_wmma<<<(KROI * 13) / 4, 128, 0, stream>>>(
      actT_sem, actT_fpn, wpk_sem, wpk_fpn, conv_sem_b, conv_fpn_b, a_f32);

  // 5) BN+ReLU on 'a' -> transposed padded bf16 [208][64]
  bnrelu_aT_kernel<<<(KROI * 208 * 64 + TPB - 1) / TPB, TPB, 0, stream>>>(
      a_f32, bn_aff_g, bn_aff_b, bn_aff_m, bn_aff_v, a_actT);

  // 6) affinity 1x1 conv + sigmoid: 256 * 169 tiles
  aff_wmma<<<(KROI * 169) / 4, 128, 0, stream>>>(wpk_aff, a_actT, conv_aff_b, Aff);

  // 7) degree normalization + fragment-major packed AsT
  degdis_kernel<<<(KROI * NODES + TPB - 1) / TPB, TPB, 0, stream>>>(Aff, dis);
  pack_ast_kernel<<<(KROI * 13 * KT_GCN * 512 + TPB - 1) / TPB, TPB, 0, stream>>>(Aff, dis, AsTpk);

  // 8) GCN layer 1
  xwT_kernel<<<(KROI * 16 * NPAD + TPB - 1) / TPB, TPB, 0, stream>>>(roi_feature, gcn1_w, XwT);
  gcn1_wmma<<<(KROI * 13) / 4, 128, 0, stream>>>(AsTpk, XwT, gcn1_b, x1b);

  // 9) GCN layer 2 (YT reuses Aff storage), LDS-staged A panel, residual accumulate
  yT_kernel<<<(KROI * CCH * NPAD + TPB - 1) / TPB, TPB, 0, stream>>>(x1b, gcn2_w, YT);
  gcn2_wmma<<<KROI * 13, 128, 0, stream>>>(AsTpk, YT, gcn2_b, out);
}
